// Model_73383811219866
// MI455X (gfx1250) — matrix operand order repaired and checked by reading, not compiled, not run on hardware
//
#include <hip/hip_runtime.h>

// ---------------------------------------------------------------------------
// CDNA5 WMMA (wave32) types: bf16 A/B fragments (16 bf16/lane = 8 VGPRs),
// f32 C/D accumulator (8 f32/lane).
// ---------------------------------------------------------------------------
typedef __attribute__((ext_vector_type(16))) __bf16 bf16x16;
typedef __attribute__((ext_vector_type(8)))  float  f32x8;
typedef __attribute__((ext_vector_type(4)))  unsigned int u32x4;

union B16Frag { bf16x16 v; u32x4 u[2]; };

#define NBLK   64      // persistent grid size (workgroups)
#define TSTEPS 512
#define HDIM   1024
#define BSZ    128
#define BH     (BSZ * HDIM)   // elements in one h state slab

// A-matrix 16x32 bf16 (MxK), row-major source, leading dim lda (elements).
// ISA: lane(0-15) row M=lane; VGPR0-3: K=k0+hi*8+[0..7]; VGPR4-7: K=k0+16+hi*8+[0..7].
__device__ __forceinline__ bf16x16 load_tile_a(const __bf16* base, int lda,
                                               int m0, int k0, int c15, int hi) {
    const __bf16* p = base + (size_t)(m0 + c15) * (size_t)lda + k0 + hi * 8;
    B16Frag r;
    r.u[0] = *(const u32x4*)(p);
    r.u[1] = *(const u32x4*)(p + 16);
    return r.v;
}

// B-matrix 32x16 bf16 (KxN) from row-major W[N_all, K] (B(k,n)=W[n][k]).
// ISA: lane(0-15) col N=lane; VGPR0-3: K=k0+hi*8+[0..7]; VGPR4-7: K=k0+16+hi*8+[0..7].
__device__ __forceinline__ bf16x16 load_tile_b(const __bf16* W, int ldw,
                                               int n0, int k0, int c15, int hi) {
    const __bf16* p = W + (size_t)(n0 + c15) * (size_t)ldw + k0 + hi * 8;
    B16Frag r;
    r.u[0] = *(const u32x4*)(p);
    r.u[1] = *(const u32x4*)(p + 16);
    return r.v;
}

__device__ __forceinline__ f32x8 wmma_bf16(bf16x16 a, bf16x16 b, f32x8 c) {
    return __builtin_amdgcn_wmma_f32_16x16x32_bf16(
        false, a, false, b, (short)0, c, false, false);
}

__device__ __forceinline__ float sigmoid_f(float x) {
    return 1.0f / (1.0f + __expf(-x));
}

// ---------------------------------------------------------------------------
// Software grid barrier (64 co-resident WGs). Each wave releases its own
// stores (per-wave store counters!) before the block barrier; one lane does
// the device-scope arrive/spin; everyone invalidates before re-reading.
// Numeric results do not depend on arrival order -> deterministic.
// ---------------------------------------------------------------------------
__device__ __forceinline__ void grid_barrier(unsigned* cnt, unsigned target) {
    __threadfence();                 // release: flush this wave's stores to device
    __syncthreads();                 // whole WG done writing
    if (threadIdx.x == 0) {
        __hip_atomic_fetch_add(cnt, 1u, __ATOMIC_RELEASE, __HIP_MEMORY_SCOPE_AGENT);
        while (__hip_atomic_load(cnt, __ATOMIC_ACQUIRE, __HIP_MEMORY_SCOPE_AGENT) < target) {
            __builtin_amdgcn_s_sleep(1);
        }
    }
    __syncthreads();
    __threadfence();                 // acquire: invalidate stale WGP$ lines
}

// ---------------------------------------------------------------------------
// Gate epilogue: c (in VGPRs) updated in place, h written as bf16.
// Lane's column j is fixed across all 8 accumulator rows.
// ---------------------------------------------------------------------------
__device__ __forceinline__ void gate_update(const f32x8& ai, const f32x8& af,
                                            const f32x8& ag, const f32x8& ao,
                                            float bi, float bf, float bg, float bo,
                                            f32x8& cacc, __bf16* hbase, size_t hoff,
                                            int m0, int hi, int j) {
#pragma unroll
    for (int r = 0; r < 8; ++r) {
        const float gi = sigmoid_f(ai[r] + bi);
        const float gf = sigmoid_f(af[r] + bf);
        const float gg = tanhf(ag[r] + bg);
        const float go = sigmoid_f(ao[r] + bo);
        const float c  = gf * cacc[r] + gi * gg;
        cacc[r] = c;
        hbase[hoff + (size_t)(m0 + hi * 8 + r) * HDIM + j] = (__bf16)(go * tanhf(c));
    }
}

// FC tile: preds[:, t, n0:n0+16] for this wave's 16 rows. K = 1024.
__device__ __forceinline__ void fc_tile(const __bf16* h, const __bf16* __restrict__ Wfc,
                                        float bb, float* out, int t,
                                        int m0, int n0, int c15, int hi) {
    f32x8 acc = {};
    for (int k0 = 0; k0 < HDIM; k0 += 32) {
        bf16x16 a = load_tile_a(h, HDIM, m0, k0, c15, hi);
        bf16x16 b = load_tile_b(Wfc, HDIM, n0, k0, c15, hi);
        acc = wmma_bf16(a, b, acc);
    }
#pragma unroll
    for (int r = 0; r < 8; ++r) {
        const int row = m0 + hi * 8 + r;   // batch
        out[((size_t)row * TSTEPS + t) * 64 + n0 + c15] = acc[r] + bb;
    }
}

// One layer's gate GEMM for this wave's (m0, n0) tile: 4 accumulator chains.
__device__ __forceinline__ void gate_gemm(const __bf16* Ax, int lda, int Kx,
                                          const __bf16* __restrict__ Wih,
                                          const __bf16* hprev,
                                          const __bf16* __restrict__ Whh,
                                          f32x8& a0, f32x8& a1, f32x8& a2, f32x8& a3,
                                          int m0, int n0, int c15, int hi) {
    for (int k0 = 0; k0 < Kx; k0 += 32) {
        bf16x16 a  = load_tile_a(Ax, lda, m0, k0, c15, hi);
        bf16x16 b0 = load_tile_b(Wih, Kx, 0 * HDIM + n0, k0, c15, hi);
        bf16x16 b1 = load_tile_b(Wih, Kx, 1 * HDIM + n0, k0, c15, hi);
        bf16x16 b2 = load_tile_b(Wih, Kx, 2 * HDIM + n0, k0, c15, hi);
        bf16x16 b3 = load_tile_b(Wih, Kx, 3 * HDIM + n0, k0, c15, hi);
        a0 = wmma_bf16(a, b0, a0);
        a1 = wmma_bf16(a, b1, a1);
        a2 = wmma_bf16(a, b2, a2);
        a3 = wmma_bf16(a, b3, a3);
    }
    for (int k0 = 0; k0 < HDIM; k0 += 32) {
        bf16x16 a  = load_tile_a(hprev, HDIM, m0, k0, c15, hi);
        bf16x16 b0 = load_tile_b(Whh, HDIM, 0 * HDIM + n0, k0, c15, hi);
        bf16x16 b1 = load_tile_b(Whh, HDIM, 1 * HDIM + n0, k0, c15, hi);
        bf16x16 b2 = load_tile_b(Whh, HDIM, 2 * HDIM + n0, k0, c15, hi);
        bf16x16 b3 = load_tile_b(Whh, HDIM, 3 * HDIM + n0, k0, c15, hi);
        a0 = wmma_bf16(a, b0, a0);
        a1 = wmma_bf16(a, b1, a1);
        a2 = wmma_bf16(a, b2, a2);
        a3 = wmma_bf16(a, b3, a3);
    }
}

// ---------------------------------------------------------------------------
// Persistent kernel: the entire T=512 stacked-LSTM + FC chain in one launch.
// Grid: 64 WGs x 256 threads. Wave w of WG b owns rows [16w,16w+16) and
// cols [16b,16b+16) of every gate block; its c-state lives in VGPRs.
// h ping-pong buffers are single slabs [2][128*1024] indexed by an integer
// offset so the compiler keeps GLOBAL (not FLAT) addressing.
// ---------------------------------------------------------------------------
__global__ __launch_bounds__(256) void lstm_persistent_kernel(
    const __bf16* __restrict__ x_bf,   // [128, 512, 64]
    const __bf16* __restrict__ Wih0,   // [4096, 64]
    const __bf16* __restrict__ Whh0,   // [4096, 1024]
    const __bf16* __restrict__ Wih1,   // [4096, 1024]
    const __bf16* __restrict__ Whh1,   // [4096, 1024]
    const __bf16* __restrict__ Wfc,    // [64, 1024]
    const float*  __restrict__ bias0,  // [4096]
    const float*  __restrict__ bias1,  // [4096]
    const float*  __restrict__ bfc,    // [64]
    __bf16* h0base,                    // [2][128,1024] layer-0 h ping-pong
    __bf16* h1base,                    // [2][128,1024] layer-1 h ping-pong
    float* out,                        // [128, 512, 64]
    unsigned* barrier_cnt)
{
    const int lane = threadIdx.x & 31;
    const int m0   = (threadIdx.x >> 5) * 16;
    const int n0   = blockIdx.x * 16;
    const int c15  = lane & 15;
    const int hi   = lane >> 4;
    const int j    = n0 + c15;

    // Biases: one scalar per lane per gate (column-indexed).
    const float b0i = bias0[0 * HDIM + j], b0f = bias0[1 * HDIM + j];
    const float b0g = bias0[2 * HDIM + j], b0o = bias0[3 * HDIM + j];
    const float b1i = bias1[0 * HDIM + j], b1f = bias1[1 * HDIM + j];
    const float b1g = bias1[2 * HDIM + j], b1o = bias1[3 * HDIM + j];
    const bool  do_fc = (blockIdx.x < 4);
    const float bfc_r = do_fc ? bfc[n0 + c15] : 0.0f;

    // Cell state lives in registers for all 512 steps.
    f32x8 c0acc = {};
    f32x8 c1acc = {};

    // Zero the h(t=0) tiles (slab 0) this wave owns, then sync the grid.
#pragma unroll
    for (int r = 0; r < 8; ++r) {
        const size_t idx = (size_t)(m0 + hi * 8 + r) * HDIM + j;
        h0base[idx] = (__bf16)0.0f;
        h1base[idx] = (__bf16)0.0f;
    }
    unsigned bar_target = 0;
    bar_target += NBLK; grid_barrier(barrier_cnt, bar_target);

    for (int t = 0; t < TSTEPS; ++t) {
        const size_t cur = (size_t)(t & 1) * BH;
        const size_t nxt = (size_t)((t & 1) ^ 1) * BH;

        // ---- Phase A: layer-0 step (+ FC of step t-1 on WGs 0..3) ----
        {
            f32x8 a0 = {}, a1 = {}, a2 = {}, a3 = {};
            gate_gemm(x_bf + (size_t)t * 64, TSTEPS * 64, 64,
                      Wih0, h0base + cur, Whh0, a0, a1, a2, a3, m0, n0, c15, hi);
            gate_update(a0, a1, a2, a3, b0i, b0f, b0g, b0o,
                        c0acc, h0base, nxt, m0, hi, j);
            if (do_fc && t > 0)
                fc_tile(h1base + cur, Wfc, bfc_r, out, t - 1, m0, n0, c15, hi);
        }
        bar_target += NBLK; grid_barrier(barrier_cnt, bar_target);

        // ---- Phase B: layer-1 step ----
        {
            f32x8 a0 = {}, a1 = {}, a2 = {}, a3 = {};
            gate_gemm(h0base + nxt, HDIM, HDIM,
                      Wih1, h1base + cur, Whh1, a0, a1, a2, a3, m0, n0, c15, hi);
            gate_update(a0, a1, a2, a3, b1i, b1f, b1g, b1o,
                        c1acc, h1base, nxt, m0, hi, j);
        }
        bar_target += NBLK; grid_barrier(barrier_cnt, bar_target);
    }

    // Final FC for t = 511 (h1 of last step is in slab (512)&1 == 0).
    if (do_fc)
        fc_tile(h1base, Wfc, bfc_r, out, TSTEPS - 1, m0, n0, c15, hi);
}

// ---------------------------------------------------------------------------
// Helpers: f32 -> bf16 conversion, bias sum, barrier-counter reset.
// ---------------------------------------------------------------------------
__global__ void cvt_bf16_kernel(const float* __restrict__ src,
                                __bf16* __restrict__ dst, int n) {
    int i = blockIdx.x * blockDim.x + threadIdx.x;
    if (i < n) dst[i] = (__bf16)src[i];
}

__global__ void bias_sum_kernel(const float* __restrict__ a,
                                const float* __restrict__ b,
                                float* __restrict__ o, int n) {
    int i = blockIdx.x * blockDim.x + threadIdx.x;
    if (i < n) o[i] = a[i] + b[i];
}

__global__ void init_counter_kernel(unsigned* cnt) {
    if (threadIdx.x == 0 && blockIdx.x == 0) *cnt = 0u;
}

// ---------------------------------------------------------------------------
// Launch: everything on `stream`, graph-capture safe (kernels only).
// ---------------------------------------------------------------------------
extern "C" void kernel_launch(void* const* d_in, const int* in_sizes, int n_in,
                              void* d_out, int out_size, void* d_ws, size_t ws_size,
                              hipStream_t stream) {
    constexpr int B = 128, T = 512, F = 64, H = 1024, OUT = 64;
    constexpr int G4H = 4 * H;

    const float* x     = (const float*)d_in[0];
    const float* W_ih0 = (const float*)d_in[1];
    const float* W_hh0 = (const float*)d_in[2];
    const float* b_ih0 = (const float*)d_in[3];
    const float* b_hh0 = (const float*)d_in[4];
    const float* W_ih1 = (const float*)d_in[5];
    const float* W_hh1 = (const float*)d_in[6];
    const float* b_ih1 = (const float*)d_in[7];
    const float* b_hh1 = (const float*)d_in[8];
    const float* W_fc  = (const float*)d_in[9];
    const float* b_fc  = (const float*)d_in[10];
    float* out = (float*)d_out;

    // Workspace carve-out (256B aligned slices).
    char*  ws  = (char*)d_ws;
    size_t off = 0;
    auto take = [&](size_t bytes) -> char* {
        char* p = ws + off;
        off += (bytes + 255) & ~(size_t)255;
        return p;
    };
    __bf16* x_bf    = (__bf16*)take((size_t)B * T * F * 2);
    __bf16* Wih0_bf = (__bf16*)take((size_t)G4H * F * 2);
    __bf16* Whh0_bf = (__bf16*)take((size_t)G4H * H * 2);
    __bf16* Wih1_bf = (__bf16*)take((size_t)G4H * H * 2);
    __bf16* Whh1_bf = (__bf16*)take((size_t)G4H * H * 2);
    __bf16* Wfc_bf  = (__bf16*)take((size_t)OUT * H * 2);
    float*  bias0   = (float*)take((size_t)G4H * 4);
    float*  bias1   = (float*)take((size_t)G4H * 4);
    __bf16* h0base  = (__bf16*)take((size_t)2 * B * H * 2);  // [2][B*H]
    __bf16* h1base  = (__bf16*)take((size_t)2 * B * H * 2);  // [2][B*H]
    unsigned* bcnt  = (unsigned*)take(256);

    auto cvt = [&](const float* s, __bf16* d, int n) {
        cvt_bf16_kernel<<<(n + 255) / 256, 256, 0, stream>>>(s, d, n);
    };
    cvt(x, x_bf, B * T * F);
    cvt(W_ih0, Wih0_bf, G4H * F);
    cvt(W_hh0, Whh0_bf, G4H * H);
    cvt(W_ih1, Wih1_bf, G4H * H);
    cvt(W_hh1, Whh1_bf, G4H * H);
    cvt(W_fc, Wfc_bf, OUT * H);
    bias_sum_kernel<<<(G4H + 255) / 256, 256, 0, stream>>>(b_ih0, b_hh0, bias0, G4H);
    bias_sum_kernel<<<(G4H + 255) / 256, 256, 0, stream>>>(b_ih1, b_hh1, bias1, G4H);
    init_counter_kernel<<<1, 32, 0, stream>>>(bcnt);

    // The whole 512-step stacked LSTM + FC in a single persistent launch.
    lstm_persistent_kernel<<<NBLK, 256, 0, stream>>>(
        x_bf, Wih0_bf, Whh0_bf, Wih1_bf, Whh1_bf, Wfc_bf,
        bias0, bias1, b_fc, h0base, h1base, out, bcnt);
}
